// AttentionBlock_12541304504581
// MI455X (gfx1250) — matrix operand order repaired and checked by reading, not compiled, not run on hardware
//
#include <hip/hip_runtime.h>
#include <hip/hip_bf16.h>

typedef __attribute__((ext_vector_type(16))) __bf16 v16bf;
typedef __attribute__((ext_vector_type(8)))  float  v8f;

#define B_   16
#define C_   512
#define NSEQ 1024
#define NH   8
#define DK   64
#define O3   1536   // 3 * NH * DK

__device__ __forceinline__ v8f wmma_bf16(v16bf a, v16bf b, v8f c) {
  return __builtin_amdgcn_wmma_f32_16x16x32_bf16(false, a, false, b, (short)0, c, false, false);
}

union F16x16 { uint4 q[2]; v16bf v; };

__device__ __forceinline__ unsigned int pack2bf(float a, float b) {
  union { __bf16 h[2]; unsigned int u; } t;
  t.h[0] = (__bf16)a; t.h[1] = (__bf16)b;
  return t.u;
}

// Async 16B global->LDS copy (CDNA5 TDM-lite path, tracked by ASYNCcnt).
// VDST = LDS byte address (low 32 bits of generic shared pointer, per ISA 10.2),
// VADDR = 64-bit global address, saddr = off.
__device__ __forceinline__ void async_copy_b128(void* lds_dst, const void* gsrc) {
  unsigned lds_off = (unsigned)(size_t)lds_dst;
  unsigned long long ga = (unsigned long long)(size_t)gsrc;
  asm volatile("global_load_async_to_lds_b128 %0, %1, off"
               :: "v"(lds_off), "v"(ga)
               : "memory");
}
__device__ __forceinline__ void wait_async0() {
  asm volatile("s_wait_asynccnt 0x0" ::: "memory");
}

// A-matrix 16x32 bf16 fragment from row-major src (ld elems/row, 16B-aligned rows).
__device__ __forceinline__ v16bf load_a_frag(const __bf16* src, int ld) {
  int lane = threadIdx.x & 31;
  const __bf16* p = src + (lane & 15) * ld + ((lane >> 4) << 3);
  F16x16 f;
  f.q[0] = *(const uint4*)(p);
  f.q[1] = *(const uint4*)(p + 16);
  return f.v;
}

// B-matrix 32x16 fragment where storage is n-major/K-contiguous: B[k][n] = src[n*ld + k].
__device__ __forceinline__ v16bf load_bT_frag(const __bf16* src, int ld) {
  int lane = threadIdx.x & 31;
  const __bf16* p = src + (lane & 15) * ld + ((lane >> 4) << 3);
  F16x16 f;
  f.q[0] = *(const uint4*)(p);
  f.q[1] = *(const uint4*)(p + 16);
  return f.v;
}

// ---------------- Kernel 1: QKV projection --------------------------------
__global__ __launch_bounds__(128) void qkv_gemm(
    const float* __restrict__ x, const float* __restrict__ wproj,
    const float* __restrict__ bproj,
    __bf16* __restrict__ q, __bf16* __restrict__ k, __bf16* __restrict__ v)
{
  __shared__ __align__(16) unsigned int At32[64 * 16];   // A[m][k] row-major, k paired
  __shared__ __align__(16) unsigned int BtT32[64 * 16];  // B^T: [n][k], k paired
  __bf16* At  = (__bf16*)At32;
  __bf16* BtT = (__bf16*)BtT32;

  int m0 = blockIdx.x * 64, n0 = blockIdx.y * 64, b = blockIdx.z;
  int t = threadIdx.x, w = t >> 5, lane = t & 31;
  int half = lane >> 4, ln = lane & 15;
  v8f acc[4] = {};

  for (int k0 = 0; k0 < C_; k0 += 32) {
    __syncthreads();
    // A tile: At[m][kk] = x[b, k0+kk, m0+m]  (transpose + cvt, paired along kk)
    for (int i = t; i < 1024; i += 128) {
      int m = i & 63, kp = i >> 6;
      float f0 = x[((size_t)b * C_ + k0 + 2 * kp)     * NSEQ + m0 + m];
      float f1 = x[((size_t)b * C_ + k0 + 2 * kp + 1) * NSEQ + m0 + m];
      At32[m * 16 + kp] = pack2bf(f0, f1);
    }
    // B tile transposed: BtT[n][kk] = wproj[k0+kk, n0+n]
    for (int i = t; i < 1024; i += 128) {
      int n = i & 63, kp = i >> 6;
      float f0 = wproj[(size_t)(k0 + 2 * kp)     * O3 + n0 + n];
      float f1 = wproj[(size_t)(k0 + 2 * kp + 1) * O3 + n0 + n];
      BtT32[n * 16 + kp] = pack2bf(f0, f1);
    }
    __syncthreads();
    v16bf af = load_a_frag(&At[(w * 16) * 32], 32);
#pragma unroll
    for (int ct = 0; ct < 4; ++ct) {
      v16bf bf = load_bT_frag(&BtT[(ct * 16) * 32], 32);
      acc[ct] = wmma_bf16(af, bf, acc[ct]);
    }
  }

#pragma unroll
  for (int ct = 0; ct < 4; ++ct) {
    int o = n0 + ct * 16 + ln;
    int head = o / 192, wi = o % 192, sel = wi / 64, d = wi % 64;
    __bf16* dst = (sel == 0) ? q : ((sel == 1) ? k : v);
    float bias = bproj[o];
#pragma unroll
    for (int r = 0; r < 8; ++r) {
      int m = m0 + w * 16 + half * 8 + r;
      dst[(((size_t)b * NH + head) * NSEQ + m) * DK + d] = (__bf16)(acc[ct][r] + bias);
    }
  }
}

// ---------------- Kernel 2: flash attention -------------------------------
__global__ __launch_bounds__(128) void attn_kernel(
    const __bf16* __restrict__ q, const __bf16* __restrict__ k,
    const __bf16* __restrict__ v, __bf16* __restrict__ res)
{
  __shared__ __align__(16) __bf16 Kt[64 * 64];           // row-major [j][d]
  __shared__ __align__(16) unsigned int VtT32[64 * 32];  // V^T: [d][j], j paired
  __shared__ __align__(16) __bf16 Pt[4][16 * 64];        // per-wave P stripe
  __bf16* VtT = (__bf16*)VtT32;

  int it = blockIdx.x, h = blockIdx.y, b = blockIdx.z;
  int t = threadIdx.x, w = t >> 5, lane = t & 31;
  int half = lane >> 4, ln = lane & 15;
  size_t bh = ((size_t)b * NH + h) * NSEQ;
  int iBase = it * 64 + w * 16;

  const __bf16* qrow = q + (bh + iBase) * DK;
  v16bf aq0 = load_a_frag(qrow, DK);
  v16bf aq1 = load_a_frag(qrow + 32, DK);

  float mrow[8], lrow[8];
  v8f o_acc[4] = {};
#pragma unroll
  for (int r = 0; r < 8; ++r) { mrow[r] = -1e30f; lrow[r] = 0.f; }

  for (int jt = 0; jt < 16; ++jt) {
    const __bf16* kbase = k + (bh + (size_t)jt * 64) * DK;
    const __bf16* vbase = v + (bh + (size_t)jt * 64) * DK;
    __syncthreads();
    {
      // K tile: async global->LDS b128 copies (ASYNCcnt), no VGPR staging
      const uint4* ksrc = (const uint4*)kbase;
      for (int i = t; i < 512; i += 128)
        async_copy_b128((uint4*)Kt + i, ksrc + i);
      // V tile: transpose into VtT[d][j] with j-pairs packed into b32 stores
      for (int i = t; i < 256; i += 128) {
        int du = i & 7, jp = i >> 3;
        uint4 r0 = *(const uint4*)(vbase + (2 * jp    ) * DK + du * 8);
        uint4 r1 = *(const uint4*)(vbase + (2 * jp + 1) * DK + du * 8);
        const unsigned int* a = (const unsigned int*)&r0;
        const unsigned int* bq = (const unsigned int*)&r1;
#pragma unroll
        for (int e = 0; e < 4; ++e) {
          unsigned int lo = (a[e] & 0xffffu) | (bq[e] << 16);
          unsigned int hi = (a[e] >> 16) | (bq[e] & 0xffff0000u);
          VtT32[(du * 8 + 2 * e    ) * 32 + jp] = lo;
          VtT32[(du * 8 + 2 * e + 1) * 32 + jp] = hi;
        }
      }
      // prefetch next tiles into cache while we compute this one
      if (jt + 1 < 16) {
        const char* kn = (const char*)(kbase + 64 * DK);
        const char* vn = (const char*)(vbase + 64 * DK);
        __builtin_prefetch(kn + t * 64, 0, 3);
        __builtin_prefetch(vn + t * 64, 0, 3);
      }
    }
    wait_async0();       // my wave's async LDS writes have landed
    __syncthreads();     // everyone's writes visible

    // S = q @ K^T
    v8f s[4] = {};
#pragma unroll
    for (int ct = 0; ct < 4; ++ct) {
      v16bf b0 = load_bT_frag(&Kt[(ct * 16) * 64], 64);
      s[ct] = wmma_bf16(aq0, b0, s[ct]);
      v16bf b1 = load_bT_frag(&Kt[(ct * 16) * 64 + 32], 64);
      s[ct] = wmma_bf16(aq1, b1, s[ct]);
    }

    // online softmax; each row lives in one 16-lane half
#pragma unroll
    for (int r = 0; r < 8; ++r) {
      float s0 = s[0][r] * 0.125f, s1 = s[1][r] * 0.125f;
      float s2 = s[2][r] * 0.125f, s3 = s[3][r] * 0.125f;
      float mx = fmaxf(fmaxf(s0, s1), fmaxf(s2, s3));
#pragma unroll
      for (int msk = 1; msk < 16; msk <<= 1)
        mx = fmaxf(mx, __shfl_xor(mx, msk, 32));
      float mnew = fmaxf(mrow[r], mx);
      float alpha = __expf(mrow[r] - mnew);
      float p0 = __expf(s0 - mnew), p1 = __expf(s1 - mnew);
      float p2 = __expf(s2 - mnew), p3 = __expf(s3 - mnew);
      float rs = p0 + p1 + p2 + p3;
#pragma unroll
      for (int msk = 1; msk < 16; msk <<= 1)
        rs += __shfl_xor(rs, msk, 32);
      lrow[r] = lrow[r] * alpha + rs;
      mrow[r] = mnew;
      s[0][r] = p0; s[1][r] = p1; s[2][r] = p2; s[3][r] = p3;
      o_acc[0][r] *= alpha; o_acc[1][r] *= alpha;
      o_acc[2][r] *= alpha; o_acc[3][r] *= alpha;
    }

    // C-layout -> row-major P stripe (per wave)
    __bf16* Pw = Pt[w];
#pragma unroll
    for (int ct = 0; ct < 4; ++ct)
#pragma unroll
      for (int r = 0; r < 8; ++r)
        Pw[(half * 8 + r) * 64 + ct * 16 + ln] = (__bf16)s[ct][r];

    // O += P @ V
#pragma unroll
    for (int kk = 0; kk < 2; ++kk) {
      v16bf ap = load_a_frag(Pw + kk * 32, 64);
#pragma unroll
      for (int ct = 0; ct < 4; ++ct) {
        v16bf bv = load_bT_frag(&VtT[(ct * 16) * 64 + kk * 32], 64);
        o_acc[ct] = wmma_bf16(ap, bv, o_acc[ct]);
      }
    }
  }

  // normalize + write res[b, i, h*64+d]
#pragma unroll
  for (int ct = 0; ct < 4; ++ct) {
    int d = ct * 16 + ln;
#pragma unroll
    for (int r = 0; r < 8; ++r) {
      int i = iBase + half * 8 + r;
      float inv = 1.0f / lrow[r];
      res[((size_t)b * NSEQ + i) * (NH * DK) + h * DK + d] = (__bf16)(o_acc[ct][r] * inv);
    }
  }
}

// ---------------- Kernel 3: output projection + residual ------------------
__global__ __launch_bounds__(128) void out_gemm(
    const __bf16* __restrict__ res, const float* __restrict__ wout,
    const float* __restrict__ bout, const float* __restrict__ x,
    float* __restrict__ out)
{
  __shared__ __align__(16) __bf16 At[64 * 32];           // A[m][k] row-major
  __shared__ __align__(16) unsigned int BtT32[64 * 16];  // B^T: [n][k], k paired
  __bf16* BtT = (__bf16*)BtT32;

  int m0 = blockIdx.x * 64, n0 = blockIdx.y * 64, b = blockIdx.z;
  int t = threadIdx.x, w = t >> 5, lane = t & 31;
  int half = lane >> 4, ln = lane & 15;
  v8f acc[4] = {};

  for (int k0 = 0; k0 < C_; k0 += 32) {
    __syncthreads();
    // A tile: async b128 copies from bf16 res (ASYNCcnt path)
    {
      uint4* At4 = (uint4*)At;
      for (int i = t; i < 256; i += 128) {
        int m = i >> 2, e = i & 3;
        async_copy_b128(At4 + i,
                        (const uint4*)(res + ((size_t)b * NSEQ + m0 + m) * C_ + k0) + e);
      }
    }
    // B tile transposed: BtT[n][kk] = wout[k0+kk, n0+n]
    for (int i = t; i < 1024; i += 128) {
      int n = i & 63, kp = i >> 6;
      float f0 = wout[(size_t)(k0 + 2 * kp)     * C_ + n0 + n];
      float f1 = wout[(size_t)(k0 + 2 * kp + 1) * C_ + n0 + n];
      BtT32[n * 16 + kp] = pack2bf(f0, f1);
    }
    wait_async0();
    __syncthreads();
    v16bf af = load_a_frag(&At[(w * 16) * 32], 32);
#pragma unroll
    for (int ct = 0; ct < 4; ++ct) {
      v16bf bf = load_bT_frag(&BtT[(ct * 16) * 32], 32);
      acc[ct] = wmma_bf16(af, bf, acc[ct]);
    }
  }

#pragma unroll
  for (int ct = 0; ct < 4; ++ct) {
    int c = n0 + ct * 16 + ln;
    float bias = bout[c];
#pragma unroll
    for (int r = 0; r < 8; ++r) {
      int n = m0 + w * 16 + half * 8 + r;
      size_t idx = ((size_t)b * C_ + c) * NSEQ + n;   // residual shares this index
      out[idx] = acc[ct][r] + bias + x[idx];
    }
  }
}

extern "C" void kernel_launch(void* const* d_in, const int* in_sizes, int n_in,
                              void* d_out, int out_size, void* d_ws, size_t ws_size,
                              hipStream_t stream) {
  const float* x     = (const float*)d_in[0];
  const float* wproj = (const float*)d_in[1];
  const float* bproj = (const float*)d_in[2];
  const float* wout  = (const float*)d_in[3];
  const float* bout  = (const float*)d_in[4];
  float* out = (float*)d_out;

  size_t qkvElems = (size_t)B_ * NH * NSEQ * DK;   // 8M elements each
  __bf16* q   = (__bf16*)d_ws;
  __bf16* k   = q + qkvElems;
  __bf16* v   = k + qkvElems;
  __bf16* res = v + qkvElems;                      // B*N*512 bf16

  qkv_gemm <<<dim3(NSEQ / 64, O3 / 64, B_), 128, 0, stream>>>(x, wproj, bproj, q, k, v);
  attn_kernel<<<dim3(NSEQ / 64, NH, B_),    128, 0, stream>>>(q, k, v, res);
  out_gemm <<<dim3(NSEQ / 64, C_ / 64, B_), 128, 0, stream>>>(res, wout, bout, x, out);
}
